// Model_68049461837945
// MI455X (gfx1250) — compile-verified, matrix operand order checked by reading, not run on hardware
//
#include <hip/hip_runtime.h>
#include <cstdint>
#include <cstddef>

// ---------------------------------------------------------------------------
// Types / helpers
// ---------------------------------------------------------------------------
typedef __bf16 bf16;
typedef bf16  v16bf __attribute__((ext_vector_type(16)));
typedef bf16  v8bf  __attribute__((ext_vector_type(8)));
typedef float v8f   __attribute__((ext_vector_type(8)));

#define DEVI __device__ __forceinline__

// Build a 16x32 (or 32x16) bf16 WMMA fragment for one lane.
// ISA 16-bit A layout: lane l (row = l&15):
//   elements 0..7  hold K = kb .. kb+7        (kb = k + 8*(l>=16))
//   elements 8..15 hold K = kb+16 .. kb+23
DEVI v16bf load_frag(const bf16* row, int kb) {
  v8bf lo = *reinterpret_cast<const v8bf*>(row + kb);
  v8bf hi = *reinterpret_cast<const v8bf*>(row + kb + 16);
  return __builtin_shufflevector(lo, hi, 0,1,2,3,4,5,6,7,8,9,10,11,12,13,14,15);
}

DEVI float gelu_exact(float x) { return 0.5f * x * (1.f + erff(x * 0.70710678118654752f)); }

// 16-byte global -> LDS async copy (ASYNCcnt path), with fallbacks.
DEVI void async_copy16(const bf16* g, bf16* l) {
#if __has_builtin(__builtin_amdgcn_global_load_async_to_lds_b128)
  typedef __attribute__((__vector_size__(4 * sizeof(int)))) int v4i_;
  __builtin_amdgcn_global_load_async_to_lds_b128(
      (__attribute__((address_space(1))) v4i_*)g,
      (__attribute__((address_space(3))) v4i_*)l, 0, 0);
#else
  unsigned la = (unsigned)(uintptr_t)l;               // generic->LDS is a truncation
  unsigned long long ga = (unsigned long long)(uintptr_t)g;
  asm volatile("global_load_async_to_lds_b128 %0, %1, off"
               :: "v"(la), "v"(ga) : "memory");
#endif
}
DEVI void wait_async0() {
#if __has_builtin(__builtin_amdgcn_s_wait_asynccnt)
  __builtin_amdgcn_s_wait_asynccnt(0);
#else
  asm volatile("s_wait_asynccnt 0x0" ::: "memory");
#endif
}

// Problem constants
#define BN   56       // bsz*nvars channel-independent rows
#define LSEQ 512      // both encoder L and decoder Ld
#define DM   512      // d_model
#define MTOT (BN*LSEQ)        // 28672 rows of the big GEMMs
#define NELT ((size_t)BN*LSEQ*DM)  // 14,680,064
#define GRID_ELT 57344        // NELT/256

// ---------------------------------------------------------------------------
// f32 -> bf16 conversion
// ---------------------------------------------------------------------------
__global__ __launch_bounds__(256) void k_f2bf(const float* __restrict__ s,
                                              bf16* __restrict__ d, int n) {
  int i = blockIdx.x * 256 + threadIdx.x;
  if (i < n) d[i] = (bf16)s[i];
}

// ---------------------------------------------------------------------------
// Generic WMMA GEMM:  C[M,N] = act( A[M,K](bf16) * W[N,K]^T(bf16) + bias + resid )
// one wave per 16x64 output strip (4 n-tiles, A fragment reused 4x)
// N must be a multiple of 64.
// ---------------------------------------------------------------------------
__global__ __launch_bounds__(32) void k_gemm(
    const bf16* __restrict__ A, const bf16* __restrict__ W,
    const float* __restrict__ bias, const float* __restrict__ resid,
    float* __restrict__ Cf, bf16* __restrict__ Ch,
    int M, int N, int K, int act)
{
  int ng = N >> 6;
  int tm = blockIdx.x / ng, tg = blockIdx.x % ng;
  int lane = threadIdx.x, half = lane >> 4, l16 = lane & 15;
  int m = tm * 16 + l16; if (m >= M) m = M - 1;
  const bf16* arow  = A + (size_t)m * K;
  const bf16* wrow0 = W + (size_t)(tg * 64 + l16) * K;
  v8f acc[4] = {};
  for (int k = 0; k < K; k += 32) {
    int kb = k + (half << 3);
    __builtin_prefetch(arow + kb + 128, 0, 1);
    __builtin_prefetch(wrow0 + kb + 128, 0, 1);
    v16bf af = load_frag(arow, kb);
    v16bf w0 = load_frag(wrow0, kb);
    v16bf w1 = load_frag(wrow0 + (size_t)16 * K, kb);
    v16bf w2 = load_frag(wrow0 + (size_t)32 * K, kb);
    v16bf w3 = load_frag(wrow0 + (size_t)48 * K, kb);
    acc[0] = __builtin_amdgcn_wmma_f32_16x16x32_bf16(false, af, false, w0, (short)0, acc[0], false, false);
    acc[1] = __builtin_amdgcn_wmma_f32_16x16x32_bf16(false, af, false, w1, (short)0, acc[1], false, false);
    acc[2] = __builtin_amdgcn_wmma_f32_16x16x32_bf16(false, af, false, w2, (short)0, acc[2], false, false);
    acc[3] = __builtin_amdgcn_wmma_f32_16x16x32_bf16(false, af, false, w3, (short)0, acc[3], false, false);
  }
#pragma unroll
  for (int q = 0; q < 4; ++q) {
    int nc = tg * 64 + q * 16 + l16;
    if (nc >= N) continue;
    float bv = bias ? bias[nc] : 0.f;
#pragma unroll
    for (int r = 0; r < 8; ++r) {
      int mr = tm * 16 + r + (half << 3);
      if (mr >= M) continue;
      float v = acc[q][r] + bv;
      if (resid) v += resid[(size_t)mr * N + nc];
      if (act) v = gelu_exact(v);
      if (Cf) Cf[(size_t)mr * N + nc] = v;
      if (Ch) Ch[(size_t)mr * N + nc] = (bf16)v;
    }
  }
}

// ---------------------------------------------------------------------------
// Autocorrelation "mean_corr" via S = Q K^T with diagonal-sum atomic epilogue
// mean_corr[b, (t-u)&511] += S[t,u]   (scale 1/512 applied in topk)
// one wave per 16x64 strip of S
// ---------------------------------------------------------------------------
__global__ __launch_bounds__(32) void k_corr(
    const bf16* __restrict__ Q, const bf16* __restrict__ Km,
    float* __restrict__ mc)
{
  int b = blockIdx.z;
  int tm = blockIdx.y, tg = blockIdx.x;
  int lane = threadIdx.x, half = lane >> 4, l16 = lane & 15;
  const bf16* qrow  = Q  + (size_t)b * LSEQ * DM + (size_t)(tm * 16 + l16) * DM;
  const bf16* krow0 = Km + (size_t)b * LSEQ * DM + (size_t)(tg * 64 + l16) * DM;
  v8f acc[4] = {};
  for (int k = 0; k < DM; k += 32) {
    int kb = k + (half << 3);
    __builtin_prefetch(qrow + kb + 128, 0, 1);
    __builtin_prefetch(krow0 + kb + 128, 0, 1);
    v16bf af = load_frag(qrow, kb);
    v16bf w0 = load_frag(krow0, kb);
    v16bf w1 = load_frag(krow0 + (size_t)16 * DM, kb);
    v16bf w2 = load_frag(krow0 + (size_t)32 * DM, kb);
    v16bf w3 = load_frag(krow0 + (size_t)48 * DM, kb);
    acc[0] = __builtin_amdgcn_wmma_f32_16x16x32_bf16(false, af, false, w0, (short)0, acc[0], false, false);
    acc[1] = __builtin_amdgcn_wmma_f32_16x16x32_bf16(false, af, false, w1, (short)0, acc[1], false, false);
    acc[2] = __builtin_amdgcn_wmma_f32_16x16x32_bf16(false, af, false, w2, (short)0, acc[2], false, false);
    acc[3] = __builtin_amdgcn_wmma_f32_16x16x32_bf16(false, af, false, w3, (short)0, acc[3], false, false);
  }
  float* mcb = mc + b * LSEQ;
#pragma unroll
  for (int q = 0; q < 4; ++q) {
    int u = tg * 64 + q * 16 + l16;
#pragma unroll
    for (int r = 0; r < 8; ++r) {
      int t = tm * 16 + r + (half << 3);
      atomicAdd(mcb + ((t - u) & 511), acc[q][r]);
    }
  }
}

// ---------------------------------------------------------------------------
// top-6 + softmax over mean_corr rows
// ---------------------------------------------------------------------------
__global__ __launch_bounds__(512) void k_topk6(const float* __restrict__ mc,
                                               float* __restrict__ tw,
                                               int* __restrict__ td)
{
  __shared__ float sv[512];
  __shared__ int   si[512];
  __shared__ float wv[6];
  __shared__ int   wd[6];
  __shared__ unsigned char taken[512];
  int b = blockIdx.x, tid = threadIdx.x;
  float myv = mc[b * 512 + tid] * (1.f / 512.f);  // mean over H*E = 512
  taken[tid] = 0;
  __syncthreads();
  for (int it = 0; it < 6; ++it) {
    sv[tid] = taken[tid] ? -1e30f : myv;
    si[tid] = tid;
    __syncthreads();
    for (int o = 256; o > 0; o >>= 1) {
      if (tid < o && sv[tid + o] > sv[tid]) { sv[tid] = sv[tid + o]; si[tid] = si[tid + o]; }
      __syncthreads();
    }
    if (tid == 0) { wv[it] = sv[0]; wd[it] = si[0]; taken[si[0]] = 1; }
    __syncthreads();
  }
  if (tid == 0) {
    float mx = wv[0];
    for (int i = 1; i < 6; ++i) mx = wv[i] > mx ? wv[i] : mx;
    float e[6], s = 0.f;
    for (int i = 0; i < 6; ++i) { e[i] = expf(wv[i] - mx); s += e[i]; }
    for (int i = 0; i < 6; ++i) { tw[b * 6 + i] = e[i] / s; td[b * 6 + i] = wd[i]; }
  }
}

// ---------------------------------------------------------------------------
// delay aggregation: out[r,t,d] = sum_i w[r,i] * V[r,(t+delay[r,i])%L, d]
// ---------------------------------------------------------------------------
__global__ __launch_bounds__(256) void k_aggregate(const bf16* __restrict__ V,
                                                   const float* __restrict__ tw,
                                                   const int* __restrict__ td,
                                                   bf16* __restrict__ Out)
{
  size_t i = (size_t)blockIdx.x * 256 + threadIdx.x;
  int d = (int)(i & 511);
  size_t rt = i >> 9;
  int t = (int)(rt & 511);
  int r = (int)(rt >> 9);
  float acc = 0.f;
#pragma unroll
  for (int k = 0; k < 6; ++k) {
    int tt = (t + td[r * 6 + k]) & 511;
    acc += tw[r * 6 + k] * (float)V[((size_t)r * 512 + tt) * 512 + d];
  }
  Out[i] = (bf16)acc;
}

// ---------------------------------------------------------------------------
// series_decomp: moving average window 25, replicate-pad (clamped indices)
// ---------------------------------------------------------------------------
__global__ __launch_bounds__(256) void k_decomp(const float* __restrict__ X,
    float* __restrict__ S, bf16* __restrict__ Sh, float* __restrict__ T, int addT)
{
  size_t i = (size_t)blockIdx.x * 256 + threadIdx.x;
  int d = (int)(i & 511);
  size_t rt = i >> 9;
  int t = (int)(rt & 511);
  size_t r = rt >> 9;
  const float* col = X + ((r << 9) << 9) + d;
  float s = 0.f;
#pragma unroll
  for (int j = -12; j <= 12; ++j) {
    int tt = t + j;
    tt = tt < 0 ? 0 : (tt > 511 ? 511 : tt);
    s += col[(size_t)tt << 9];
  }
  float ma = s * (1.f / 25.f);
  float se = col[(size_t)t << 9] - ma;
  if (S)  S[i] = se;
  if (Sh) Sh[i] = (bf16)se;
  if (T) { if (addT) T[i] += ma; else T[i] = ma; }
}

// ---------------------------------------------------------------------------
// embedding: circular conv3 (Cin=1) + time-feature embedding
// ---------------------------------------------------------------------------
__global__ __launch_bounds__(256) void k_embed(const float* __restrict__ xe,
    const float* __restrict__ mark, const float* __restrict__ cw,
    const float* __restrict__ twt, float* __restrict__ Fo, bf16* __restrict__ Ho)
{
  size_t i = (size_t)blockIdx.x * 256 + threadIdx.x;
  int o = (int)(i & 511);
  size_t rt = i >> 9;
  int t = (int)(rt & 511);
  int r = (int)(rt >> 9);
  int b = r / 7;
  const float* xr = xe + (size_t)r * 512;
  float xm = xr[(t + 511) & 511], x0 = xr[t], xp = xr[(t + 1) & 511];
  float v = cw[o * 3] * xm + cw[o * 3 + 1] * x0 + cw[o * 3 + 2] * xp;
  const float* mk = mark + ((size_t)b * 512 + t) * 4;
  v += twt[o * 4] * mk[0] + twt[o * 4 + 1] * mk[1] + twt[o * 4 + 2] * mk[2] + twt[o * 4 + 3] * mk[3];
  Fo[i] = v;
  Ho[i] = (bf16)v;
}

// ---------------------------------------------------------------------------
// Fused FFN: Out = resid + gelu(X * W1^T) * W2^T        (per 16-row tile)
// A tile staged in LDS via async copies; hidden kept in LDS as bf16,
// processed in two 1024-wide K chunks
// ---------------------------------------------------------------------------
__global__ __launch_bounds__(256) void k_ffn(
    const bf16* __restrict__ X, const bf16* __restrict__ W1,
    const bf16* __restrict__ W2, const float* __restrict__ resid,
    float* __restrict__ Out, int M)
{
  const int LDA = 520, LDH = 1032;
  __shared__ bf16 sA[16 * 520];
  __shared__ bf16 sH[16 * 1032];
  int tm = blockIdx.x;
  int tid = threadIdx.x, wave = tid >> 5, lane = tid & 31;
  int half = lane >> 4, l16 = lane & 15;
  // stage the 16x512 A tile in LDS via async global->LDS copies
  for (int i = tid; i < 16 * 64; i += 256) {
    int rr = i >> 6, cc = (i & 63) << 3;
    int mr = tm * 16 + rr; if (mr >= M) mr = M - 1;
    async_copy16(X + (size_t)mr * 512 + cc, sA + rr * LDA + cc);
  }
  wait_async0();
  __syncthreads();
  v8f acc[4] = {};
  for (int ch = 0; ch < 2; ++ch) {
    // phase 1: hidden chunk (16 x 1024), tiles nt in [ch*64, ch*64+64)
    for (int q = 0; q < 8; ++q) {
      int nt = ch * 64 + wave * 8 + q;
      const bf16* wrow = W1 + (size_t)(nt * 16 + l16) * 512;
      const bf16* arow = sA + l16 * LDA;
      v8f h = {};
      for (int k = 0; k < 512; k += 32) {
        int kb = k + (half << 3);
        __builtin_prefetch(wrow + kb + 128, 0, 1);
        v16bf af = load_frag(arow, kb);
        v16bf wf = load_frag(wrow, kb);
        h = __builtin_amdgcn_wmma_f32_16x16x32_bf16(false, af, false, wf,
                                                    (short)0, h, false, false);
      }
      int col = (nt - ch * 64) * 16 + l16;
#pragma unroll
      for (int r2 = 0; r2 < 8; ++r2)
        sH[(r2 + (half << 3)) * LDH + col] = (bf16)gelu_exact(h[r2]);
    }
    __syncthreads();
    // phase 2 (partial over this K chunk): output tiles n2 = wave*4+q
    for (int q = 0; q < 4; ++q) {
      int n2 = wave * 4 + q;
      const bf16* wrow = W2 + (size_t)(n2 * 16 + l16) * 2048 + ch * 1024;
      const bf16* hrow = sH + l16 * LDH;
      for (int k = 0; k < 1024; k += 32) {
        int kb = k + (half << 3);
        __builtin_prefetch(wrow + kb + 128, 0, 1);
        v16bf hf = load_frag(hrow, kb);
        v16bf wf = load_frag(wrow, kb);
        acc[q] = __builtin_amdgcn_wmma_f32_16x16x32_bf16(false, hf, false, wf,
                                                         (short)0, acc[q], false, false);
      }
    }
    __syncthreads();
  }
  for (int q = 0; q < 4; ++q) {
    int nc = (wave * 4 + q) * 16 + l16;
#pragma unroll
    for (int r2 = 0; r2 < 8; ++r2) {
      int mr = tm * 16 + r2 + (half << 3);
      if (mr < M) {
        size_t o = (size_t)mr * 512 + nc;
        Out[o] = resid[o] + acc[q][r2];
      }
    }
  }
}

// ---------------------------------------------------------------------------
// my_layernorm pieces
// ---------------------------------------------------------------------------
__global__ __launch_bounds__(256) void k_ln_rows(const float* __restrict__ X,
    const float* __restrict__ w, const float* __restrict__ b, float* __restrict__ Y)
{
  __shared__ float red[256];
  size_t row = blockIdx.x;
  const float* x = X + row * 512;
  int tid = threadIdx.x;
  float a0 = x[tid], a1 = x[tid + 256];
  red[tid] = a0 + a1;
  __syncthreads();
  for (int o = 128; o > 0; o >>= 1) { if (tid < o) red[tid] += red[tid + o]; __syncthreads(); }
  float mu = red[0] * (1.f / 512.f);
  __syncthreads();
  float d0 = a0 - mu, d1 = a1 - mu;
  red[tid] = d0 * d0 + d1 * d1;
  __syncthreads();
  for (int o = 128; o > 0; o >>= 1) { if (tid < o) red[tid] += red[tid + o]; __syncthreads(); }
  float inv = rsqrtf(red[0] * (1.f / 512.f) + 1e-5f);
  Y[row * 512 + tid]       = d0 * inv * w[tid] + b[tid];
  Y[row * 512 + tid + 256] = d1 * inv * w[tid + 256] + b[tid + 256];
}

__global__ __launch_bounds__(256) void k_colmean(const float* __restrict__ Y,
                                                 float* __restrict__ cm)
{
  int i = blockIdx.x * 256 + threadIdx.x;
  if (i >= BN * 512) return;
  int r = i >> 9, d = i & 511;
  const float* p = Y + (((size_t)r << 9) << 9) + d;
  float s = 0.f;
  for (int t = 0; t < 512; ++t) s += p[(size_t)t << 9];
  cm[i] = s * (1.f / 512.f);
}

__global__ __launch_bounds__(256) void k_sub_cm(const float* __restrict__ Y,
    const float* __restrict__ cm, float* __restrict__ Fo, bf16* __restrict__ Ho)
{
  size_t i = (size_t)blockIdx.x * 256 + threadIdx.x;
  int d = (int)(i & 511);
  size_t rt = i >> 9;
  int r = (int)(rt >> 9);
  float v = Y[i] - cm[(r << 9) + d];
  if (Fo) Fo[i] = v;
  if (Ho) Ho[i] = (bf16)v;
}

// ---------------------------------------------------------------------------
// decoder init: seasonal_init rows + trend7 init (broadcast trend/mean)
// ---------------------------------------------------------------------------
__global__ __launch_bounds__(256) void k_dec_init(const float* __restrict__ xe,
    float* __restrict__ sbuf, float* __restrict__ t7)
{
  int u = blockIdx.x * 256 + threadIdx.x;
  if (u >= BN * 512) return;
  int r = u >> 9, t = u & 511;
  const float* xr = xe + (size_t)r * 512;
  float sval, tval;
  if (t < 256) {
    int p = 256 + t;
    float s = 0.f;
#pragma unroll
    for (int j = -12; j <= 12; ++j) {
      int tt = p + j; tt = tt > 511 ? 511 : tt;
      s += xr[tt];
    }
    float ma = s * (1.f / 25.f);
    sval = xr[p] - ma;
    tval = ma;
  } else {
    float s = 0.f;
    for (int tt = 0; tt < 512; ++tt) s += xr[tt];
    sval = 0.f;
    tval = s * (1.f / 512.f);
  }
  sbuf[u] = sval;
  for (int c = 0; c < 7; ++c) t7[(size_t)u * 7 + c] = tval;
}

// ---------------------------------------------------------------------------
// trend projection conv: trend7 += circ_conv3(Tacc, trend_w)   (C_OUT=7)
// ---------------------------------------------------------------------------
__global__ __launch_bounds__(256) void k_trend_conv(const float* __restrict__ Tacc,
    const float* __restrict__ tw3, float* __restrict__ t7)
{
  int i = blockIdx.x * 256 + threadIdx.x;
  if (i >= BN * 512 * 7) return;
  int c = i % 7;
  int ru = i / 7;
  int u = ru & 511, r = ru >> 9;
  const float* r0 = Tacc + ((size_t)r * 512 + ((u + 511) & 511)) * 512;
  const float* r1 = Tacc + ((size_t)r * 512 + u) * 512;
  const float* r2 = Tacc + ((size_t)r * 512 + ((u + 1) & 511)) * 512;
  const float* w = tw3 + (size_t)c * 512 * 3;
  float acc = 0.f;
  for (int d = 0; d < 512; ++d)
    acc += r0[d] * w[d * 3] + r1[d] * w[d * 3 + 1] + r2[d] * w[d * 3 + 2];
  t7[i] += acc;
}

// ---------------------------------------------------------------------------
// seasonal projection (d_model -> 1) + dec_out flatten to bf16 (56 x 3584)
// ---------------------------------------------------------------------------
__global__ __launch_bounds__(256) void k_season(const float* __restrict__ X,
    const float* __restrict__ w, const float* __restrict__ b,
    const float* __restrict__ t7, bf16* __restrict__ df)
{
  int u = blockIdx.x * 256 + threadIdx.x;
  if (u >= BN * 512) return;
  const float* x = X + (size_t)u * 512;
  float s = b[0];
  for (int d = 0; d < 512; ++d) s += x[d] * w[d];
  for (int c = 0; c < 7; ++c)
    df[(size_t)u * 7 + c] = (bf16)(t7[(size_t)u * 7 + c] + s);
}

// final: (8,7,256) -> (8,256,7)
__global__ __launch_bounds__(256) void k_final(const float* __restrict__ ho,
                                               float* __restrict__ out)
{
  int i = blockIdx.x * 256 + threadIdx.x;
  if (i >= 8 * 256 * 7) return;
  int v = i % 7, bt = i / 7, t = bt % 256, b = bt / 256;
  out[i] = ho[(b * 7 + v) * 256 + t];
}

// ---------------------------------------------------------------------------
// Host-side orchestration
// ---------------------------------------------------------------------------
struct AttnW { const bf16 *wq, *wk, *wv, *wo; const float *bq, *bk, *bv, *bo; };

static void run_gemm(const bf16* A, const bf16* W, const float* bias,
                     const float* resid, float* Cf, bf16* Ch,
                     int M, int N, int K, int act, hipStream_t s)
{
  int tiles = ((M + 15) / 16) * (N / 64);
  k_gemm<<<tiles, 32, 0, s>>>(A, W, bias, resid, Cf, Ch, M, N, K, act);
}

static void run_attn(const float* xres, const bf16* hx, const bf16* hkv,
                     const AttnW& a, float* xout,
                     bf16* Hq, bf16* Hk, bf16* Hv, bf16* Hg,
                     float* MC, float* TW, int* TD, hipStream_t s)
{
  run_gemm(hx,  a.wq, a.bq, nullptr, nullptr, Hq, MTOT, DM, DM, 0, s);
  run_gemm(hkv, a.wk, a.bk, nullptr, nullptr, Hk, MTOT, DM, DM, 0, s);
  run_gemm(hkv, a.wv, a.bv, nullptr, nullptr, Hv, MTOT, DM, DM, 0, s);
  (void)hipMemsetAsync(MC, 0, (size_t)BN * 512 * 4, s);
  k_corr<<<dim3(8, 32, BN), 32, 0, s>>>(Hq, Hk, MC);
  k_topk6<<<BN, 512, 0, s>>>(MC, TW, TD);
  k_aggregate<<<GRID_ELT, 256, 0, s>>>(Hv, TW, TD, Hg);
  run_gemm(Hg, a.wo, a.bo, xres, xout, nullptr, MTOT, DM, DM, 0, s);
}

extern "C" void kernel_launch(void* const* d_in, const int* in_sizes, int n_in,
                              void* d_out, int out_size, void* d_ws, size_t ws_size,
                              hipStream_t stream)
{
  (void)in_sizes; (void)n_in; (void)out_size; (void)ws_size;
  const float* x_enc    = (const float*)d_in[0];
  const float* mark_enc = (const float*)d_in[1];
  const float* mark_dec = (const float*)d_in[3];
  const float* enc_cw   = (const float*)d_in[4];
  const float* enc_tw   = (const float*)d_in[5];
  const float* dec_cw   = (const float*)d_in[6];
  const float* dec_tw   = (const float*)d_in[7];

  char* w = (char*)d_ws;
  const size_t SZF = (size_t)NELT * 4, SZH = (size_t)NELT * 2;
  float* F0 = (float*)(w);
  float* F1 = (float*)(w + SZF);
  float* F2 = (float*)(w + 2 * SZF);          // decoder trend accumulator
  bf16*  Hx = (bf16*)(w + 3 * SZF);
  bf16*  Hq = (bf16*)(w + 3 * SZF + SZH);
  bf16*  Hk = (bf16*)(w + 3 * SZF + 2 * SZH);
  bf16*  Hv = (bf16*)(w + 3 * SZF + 3 * SZH);
  bf16*  He = (bf16*)(w + 3 * SZF + 4 * SZH);
  bf16*  Hg = Hq;                              // Q dead by aggregation time
  char*  WT = w + 3 * SZF + 5 * SZH;           // bf16 weight arena (~23 MB)

  size_t woff = 0;
  auto cvt = [&](int idx, size_t n) -> bf16* {
    bf16* dst = (bf16*)(WT + woff);
    woff += n * 2;
    k_f2bf<<<(int)((n + 255) / 256), 256, 0, stream>>>((const float*)d_in[idx], dst, (int)n);
    return dst;
  };
  auto attn_of = [&](int base) -> AttnW {
    AttnW a;
    a.wq = cvt(base + 0, 262144); a.bq = (const float*)d_in[base + 1];
    a.wk = cvt(base + 2, 262144); a.bk = (const float*)d_in[base + 3];
    a.wv = cvt(base + 4, 262144); a.bv = (const float*)d_in[base + 5];
    a.wo = cvt(base + 6, 262144); a.bo = (const float*)d_in[base + 7];
    return a;
  };
  AttnW enc0 = attn_of(8);
  bf16* W1e0 = cvt(16, 1048576); bf16* W2e0 = cvt(17, 1048576);
  AttnW enc1 = attn_of(18);
  bf16* W1e1 = cvt(26, 1048576); bf16* W2e1 = cvt(27, 1048576);
  const float* enc_nw = (const float*)d_in[28];
  const float* enc_nb = (const float*)d_in[29];
  AttnW dself = attn_of(30);
  AttnW dcross = attn_of(38);
  bf16* W1d = cvt(46, 1048576); bf16* W2d = cvt(47, 1048576);
  const float* trend_w = (const float*)d_in[48];
  const float* dec_nw = (const float*)d_in[49];
  const float* dec_nb = (const float*)d_in[50];
  const float* proj_w = (const float*)d_in[51];
  const float* proj_b = (const float*)d_in[52];
  bf16* HeadW = cvt(53, 917504);
  const float* head_b = (const float*)d_in[54];

  char* SM = WT + 24ull * 1024 * 1024;
  size_t soff = 0;
  auto salloc = [&](size_t bytes) -> char* {
    char* p = SM + soff; soff += (bytes + 255) & ~255ull; return p;
  };
  float* MC   = (float*)salloc((size_t)BN * 512 * 4);
  float* TW   = (float*)salloc((size_t)BN * 6 * 4);
  int*   TD   = (int*)  salloc((size_t)BN * 6 * 4);
  float* SBuf = (float*)salloc((size_t)BN * 512 * 4);
  float* T7   = (float*)salloc((size_t)BN * 512 * 7 * 4);
  float* CM   = (float*)salloc((size_t)BN * 512 * 4);
  bf16*  DF   = (bf16*) salloc((size_t)BN * 3584 * 2);
  float* HO   = (float*)salloc((size_t)BN * 256 * 4);

  // ---------------- encoder ----------------
  k_embed<<<GRID_ELT, 256, 0, stream>>>(x_enc, mark_enc, enc_cw, enc_tw, F0, Hx);
  const AttnW* eatt[2] = { &enc0, &enc1 };
  bf16* ew1[2] = { W1e0, W1e1 };
  bf16* ew2[2] = { W2e0, W2e1 };
  for (int l = 0; l < 2; ++l) {
    run_attn(F0, Hx, Hx, *eatt[l], F1, Hq, Hk, Hv, Hg, MC, TW, TD, stream);
    k_decomp<<<GRID_ELT, 256, 0, stream>>>(F1, F0, Hx, nullptr, 0);
    k_ffn<<<MTOT / 16, 256, 0, stream>>>(Hx, ew1[l], ew2[l], F0, F1, MTOT);
    k_decomp<<<GRID_ELT, 256, 0, stream>>>(F1, F0, Hx, nullptr, 0);
  }
  k_ln_rows<<<MTOT, 256, 0, stream>>>(F0, enc_nw, enc_nb, F1);
  k_colmean<<<112, 256, 0, stream>>>(F1, CM);
  k_sub_cm<<<GRID_ELT, 256, 0, stream>>>(F1, CM, nullptr, He);   // enc_out -> bf16

  // ---------------- decoder ----------------
  k_dec_init<<<112, 256, 0, stream>>>(x_enc, SBuf, T7);
  k_embed<<<GRID_ELT, 256, 0, stream>>>(SBuf, mark_dec, dec_cw, dec_tw, F0, Hx);

  run_attn(F0, Hx, Hx, dself, F1, Hq, Hk, Hv, Hg, MC, TW, TD, stream);
  k_decomp<<<GRID_ELT, 256, 0, stream>>>(F1, F0, Hx, F2, 0);      // t1
  run_attn(F0, Hx, He, dcross, F1, Hq, Hk, Hv, Hg, MC, TW, TD, stream);
  k_decomp<<<GRID_ELT, 256, 0, stream>>>(F1, F0, Hx, F2, 1);      // +t2
  k_ffn<<<MTOT / 16, 256, 0, stream>>>(Hx, W1d, W2d, F0, F1, MTOT);
  k_decomp<<<GRID_ELT, 256, 0, stream>>>(F1, F0, Hx, F2, 1);      // +t3
  k_trend_conv<<<(BN * 512 * 7 + 255) / 256, 256, 0, stream>>>(F2, trend_w, T7);

  k_ln_rows<<<MTOT, 256, 0, stream>>>(F0, dec_nw, dec_nb, F1);
  k_colmean<<<112, 256, 0, stream>>>(F1, CM);
  k_sub_cm<<<GRID_ELT, 256, 0, stream>>>(F1, CM, F0, nullptr);
  k_season<<<112, 256, 0, stream>>>(F0, proj_w, proj_b, T7, DF);

  run_gemm(DF, HeadW, head_b, nullptr, HO, nullptr, BN, 256, 3584, 0, stream);
  k_final<<<(8 * 256 * 7 + 255) / 256, 256, 0, stream>>>(HO, (float*)d_out);
}
